// SparseResBlockC2S3d_44933947851039
// MI455X (gfx1250) — matrix operand order repaired and checked by reading, not compiled, not run on hardware
//
#include <hip/hip_runtime.h>
#include <cstdint>

// ---------------- CDNA5 WMMA plumbing (gfx1250, wave32) ----------------
typedef __bf16 v16bf __attribute__((ext_vector_type(16)));
typedef float  v8f   __attribute__((ext_vector_type(8)));

union ABFrag  { v16bf v; uint32_t u[8]; uint4 q[2]; };
union AccFrag { v8f   v; float f[8]; };

__device__ __forceinline__ uint16_t f2bf(float x) {
  uint32_t u = __float_as_uint(x);
  u += 0x7FFFu + ((u >> 16) & 1u);      // round-to-nearest-even
  return (uint16_t)(u >> 16);
}

// A-matrix 16x32 bf16 fragment (ISA 7.12.2): lanes 0-15 -> M=0..15, K=kk..kk+7 & kk+16..kk+23
// lanes 16-31 -> same M, K=kk+8..kk+15 & kk+24..kk+31.  Source tile is row-major [16 x rowstride].
__device__ __forceinline__ void load_a_frag(ABFrag& fr, const uint16_t* base, int rowstride, int kk) {
  int lane = threadIdx.x & 31;
  const uint16_t* p = base + (lane & 15) * rowstride + kk + ((lane >> 4) << 3);
  fr.q[0] = *(const uint4*)p;
  fr.q[1] = *(const uint4*)(p + 16);
}

// B-matrix 32x16 bf16 fragment: lane holds column N=lane&15; lanes 0-15 K=kk..kk+7 & kk+16..kk+23,
// lanes 16-31 K=kk+8..kk+15 & kk+24..kk+31 (same K order as A). Source rows are [co][K contiguous].
__device__ __forceinline__ void load_b_frag(ABFrag& fr, const uint16_t* base, int rowstride, int kk) {
  int lane = threadIdx.x & 31;
  const uint16_t* p = base + (lane & 15) * rowstride + kk + ((lane >> 4) << 3);
  fr.q[0] = *(const uint4*)p;
  fr.q[1] = *(const uint4*)(p + 16);
}

__device__ __forceinline__ void wmma_acc(AccFrag& acc, const ABFrag& a, const ABFrag& b) {
  acc.v = __builtin_amdgcn_wmma_f32_16x16x32_bf16(false, a.v, false, b.v, (short)0, acc.v,
                                                  false, false);
}

// ---------------- problem constants ----------------
constexpr int N   = 16384;
constexpr int C   = 128;
constexpr int CO  = 128;
constexpr int GD  = 64;         // coarse grid
constexpr int GF  = 128;        // fine grid
constexpr int NF  = N * 8;      // fine voxels
constexpr int WSTRIDE = 27 * C; // weight row stride in elements ([Co][27][Ci])
constexpr int MT  = 32;         // points per workgroup (M tile)
constexpr float EPS = 1e-6f;

// ---------------- helpers ----------------
__device__ __forceinline__ float blockReduceSum128(float v, float* red) {
  int t = threadIdx.x;
  red[t] = v; __syncthreads();
  #pragma unroll
  for (int s = 64; s > 0; s >>= 1) { if (t < s) red[t] += red[t + s]; __syncthreads(); }
  float r = red[0]; __syncthreads();
  return r;
}

// ---------------- kernels ----------------
__global__ __launch_bounds__(128) void build_tables_k(const int* __restrict__ coords,
                                                      int* __restrict__ tc, int* __restrict__ tf) {
  int p = blockIdx.x * 128 + threadIdx.x;
  if (p >= N) return;
  int sx = coords[p * 4 + 1], sy = coords[p * 4 + 2], sz = coords[p * 4 + 3];
  tc[(sx * GD + sy) * GD + sz] = p;
  #pragma unroll
  for (int j = 0; j < 8; ++j) {
    int fx = 2 * sx + (j & 1), fy = 2 * sy + ((j >> 1) & 1), fz = 2 * sz + ((j >> 2) & 1);
    tf[(fx * GF + fy) * GF + fz] = p * 8 + j;
  }
}

__global__ __launch_bounds__(256) void cvt_bf16_k(const float* __restrict__ src,
                                                  uint16_t* __restrict__ dst, int n) {
  int i = blockIdx.x * 256 + threadIdx.x;
  if (i < n) dst[i] = f2bf(src[i]);
}

// subdiv / amask / h = silu(LN(x)) in bf16.  One block of 128 threads per point.
__global__ __launch_bounds__(128) void preprocess_k(const float* __restrict__ x,
                                                    const float* __restrict__ gamma,
                                                    const float* __restrict__ beta,
                                                    const float* __restrict__ w_sub,
                                                    const float* __restrict__ b_sub,
                                                    uint16_t* __restrict__ h_bf,
                                                    float* __restrict__ amask,
                                                    float* __restrict__ subdiv) {
  __shared__ float red[128];
  int p = blockIdx.x, t = threadIdx.x;
  float v = x[(size_t)p * C + t];
  float mean = blockReduceSum128(v, red) * (1.f / C);
  float xc = v - mean;
  float var = blockReduceSum128(xc * xc, red) * (1.f / C);
  float y = xc * rsqrtf(var + EPS);
  y = y * gamma[t] + beta[t];
  float h = y / (1.f + __expf(-y));             // silu
  h_bf[(size_t)p * C + t] = f2bf(h);
  #pragma unroll
  for (int j = 0; j < 8; ++j) {
    float s = blockReduceSum128(v * w_sub[j * C + t], red);
    if (t == 0) {
      s += b_sub[j];
      subdiv[p * 8 + j] = s;
      amask[p * 8 + j] = (s > 0.f) ? 1.f : 0.f;
    }
  }
}

// Gather a 32-row x 128-col bf16 A tile from src rows nbr[] (zero rows where nbr<0).
__device__ __forceinline__ void gather_tile32(uint16_t* Atile, const uint16_t* src,
                                              const int* nbr) {
  int t = threadIdx.x;
  int row = t >> 2, col = (t & 3) << 5;        // 32 halfs (64B) per thread
  int nb = nbr[row];
  uint4 z0 = {0,0,0,0}, z1 = z0, z2 = z0, z3 = z0;
  if (nb >= 0) {
    const uint4* s = (const uint4*)(src + (size_t)nb * 128 + col);
    z0 = s[0]; z1 = s[1]; z2 = s[2]; z3 = s[3];
  }
  uint4* d = (uint4*)&Atile[row * 128 + col];
  d[0] = z0; d[1] = z1; d[2] = z2; d[3] = z3;
}

// conv1: WG = 32 coarse points x 128 out channels (child slice j = blockIdx.y)
// fused: bias -> amask -> LN -> silu -> bf16 (input of conv2)
__global__ __launch_bounds__(128) void conv1_k(const uint16_t* __restrict__ h_bf,
                                               const int* __restrict__ coords,
                                               const int* __restrict__ tc,
                                               const uint16_t* __restrict__ w1b,
                                               const float* __restrict__ b1,
                                               const float* __restrict__ amask,
                                               uint16_t* __restrict__ h2in) {
  __shared__ __align__(16) uint16_t Atile[MT * 128];     // 8 KB
  __shared__ int nbr[MT], sx[MT], sy[MT], sz[MT];
  __shared__ float outS[MT][128];                        // 16 KB
  __shared__ float rmean[MT], rrstd[MT];

  int tileP = blockIdx.x, j = blockIdx.y;
  int t = threadIdx.x, wave = t >> 5, lane = t & 31;
  int cbase = wave * 32;                                 // wave's out-channel base

  if (t < MT) {
    int p = tileP * MT + t;
    sx[t] = coords[p * 4 + 1]; sy[t] = coords[p * 4 + 2]; sz[t] = coords[p * 4 + 3];
  }
  __syncthreads();

  AccFrag a00, a01, a10, a11;                            // [mtile][ntile]
  #pragma unroll
  for (int r = 0; r < 8; ++r) { a00.f[r]=0.f; a01.f[r]=0.f; a10.f[r]=0.f; a11.f[r]=0.f; }

  const uint16_t* wcol = w1b + (size_t)(j * 128) * WSTRIDE;   // this block's 128 co rows

  for (int k = 0; k < 27; ++k) {
    int d0 = k / 9 - 1, d1 = (k / 3) % 3 - 1, d2 = k % 3 - 1;
    if (t < MT) {
      int nx = sx[t] + d0, ny = sy[t] + d1, nz = sz[t] + d2;
      int nb = -1;
      if ((unsigned)nx < (unsigned)GD && (unsigned)ny < (unsigned)GD && (unsigned)nz < (unsigned)GD)
        nb = tc[(nx * GD + ny) * GD + nz];
      nbr[t] = nb;
    }
    // prefetch next offset's weight rows (one thread per co row, 2 lines per row)
    if (k + 1 < 27) {
      const uint16_t* wnext = wcol + (size_t)t * WSTRIDE + (k + 1) * C;
      __builtin_prefetch(wnext, 0, 1);
      __builtin_prefetch(wnext + 64, 0, 1);
    }
    __syncthreads();
    gather_tile32(Atile, h_bf, nbr);
    __syncthreads();

    const uint16_t* wb = wcol + (size_t)cbase * WSTRIDE + k * C;
    #pragma unroll
    for (int kk = 0; kk < C; kk += 32) {
      ABFrag fa0, fa1, fb0, fb1;
      load_a_frag(fa0, Atile, 128, kk);
      load_a_frag(fa1, Atile + 16 * 128, 128, kk);
      load_b_frag(fb0, wb, WSTRIDE, kk);
      load_b_frag(fb1, wb + (size_t)16 * WSTRIDE, WSTRIDE, kk);
      wmma_acc(a00, fa0, fb0);
      wmma_acc(a01, fa0, fb1);
      wmma_acc(a10, fa1, fb0);
      wmma_acc(a11, fa1, fb1);
    }
    __syncthreads();
  }

  // scatter accumulators: D layout — VGPR r, lane l: M = r + 8*(l>>4), N = l&15
  {
    int n0 = cbase + (lane & 15);
    int mo = (lane >> 4) * 8;
    #pragma unroll
    for (int r = 0; r < 8; ++r) {
      outS[r + mo][n0]           = a00.f[r];
      outS[r + mo][n0 + 16]      = a01.f[r];
      outS[r + mo + 16][n0]      = a10.f[r];
      outS[r + mo + 16][n0 + 16] = a11.f[r];
    }
  }
  __syncthreads();

  // bias + mask + row stats (LN over the 128 channels of each fine row)
  if (t < MT) {
    int f = (tileP * MT + t) * 8 + j;
    float msk = amask[f];
    float s = 0.f, ss = 0.f;
    for (int c = 0; c < 128; ++c) {
      float v = (outS[t][c] + b1[j * 128 + c]) * msk;
      outS[t][c] = v;
      s += v; ss += v * v;
    }
    float mean = s * (1.f / 128.f);
    float var = ss * (1.f / 128.f) - mean * mean;
    rmean[t] = mean;
    rrstd[t] = rsqrtf(fmaxf(var, 0.f) + EPS);
  }
  __syncthreads();

  {   // normalize + silu -> bf16 conv2 input; 32 elems per thread
    int row = t >> 2, c0 = (t & 3) << 5;
    int f = (tileP * MT + row) * 8 + j;
    float mean = rmean[row], rstd = rrstd[row];
    #pragma unroll
    for (int c = c0; c < c0 + 32; ++c) {
      float y = (outS[row][c] - mean) * rstd;
      float sv = y / (1.f + __expf(-y));
      h2in[(size_t)f * CO + c] = f2bf(sv);
    }
  }
}

// conv2 on the fine grid: WG = 32 fine voxels x 128 out channels; epilogue writes d_out
__global__ __launch_bounds__(128) void conv2_k(const uint16_t* __restrict__ h2in,
                                               const int* __restrict__ coords,
                                               const int* __restrict__ tf,
                                               const uint16_t* __restrict__ w2b,
                                               const float* __restrict__ b2,
                                               const float* __restrict__ amask,
                                               const float* __restrict__ x,
                                               float* __restrict__ out) {
  __shared__ __align__(16) uint16_t Atile[MT * 128];     // 8 KB
  __shared__ int nbr[MT], sx[MT], sy[MT], sz[MT];

  int tileF = blockIdx.x;
  int t = threadIdx.x, wave = t >> 5, lane = t & 31;
  int cbase = wave * 32;

  if (t < MT) {
    int f = tileF * MT + t;
    int p = f >> 3, j = f & 7;
    sx[t] = 2 * coords[p * 4 + 1] + (j & 1);
    sy[t] = 2 * coords[p * 4 + 2] + ((j >> 1) & 1);
    sz[t] = 2 * coords[p * 4 + 3] + ((j >> 2) & 1);
  }
  __syncthreads();

  AccFrag a00, a01, a10, a11;
  #pragma unroll
  for (int r = 0; r < 8; ++r) { a00.f[r]=0.f; a01.f[r]=0.f; a10.f[r]=0.f; a11.f[r]=0.f; }

  for (int k = 0; k < 27; ++k) {
    int d0 = k / 9 - 1, d1 = (k / 3) % 3 - 1, d2 = k % 3 - 1;
    if (t < MT) {
      int nx = sx[t] + d0, ny = sy[t] + d1, nz = sz[t] + d2;
      int nb = -1;
      if ((unsigned)nx < (unsigned)GF && (unsigned)ny < (unsigned)GF && (unsigned)nz < (unsigned)GF)
        nb = tf[(nx * GF + ny) * GF + nz];
      nbr[t] = nb;
    }
    if (k + 1 < 27) {
      const uint16_t* wnext = w2b + (size_t)t * WSTRIDE + (k + 1) * CO;
      __builtin_prefetch(wnext, 0, 1);
      __builtin_prefetch(wnext + 64, 0, 1);
    }
    __syncthreads();
    gather_tile32(Atile, h2in, nbr);
    __syncthreads();

    const uint16_t* wb = w2b + (size_t)cbase * WSTRIDE + k * CO;
    #pragma unroll
    for (int kk = 0; kk < CO; kk += 32) {
      ABFrag fa0, fa1, fb0, fb1;
      load_a_frag(fa0, Atile, 128, kk);
      load_a_frag(fa1, Atile + 16 * 128, 128, kk);
      load_b_frag(fb0, wb, WSTRIDE, kk);
      load_b_frag(fb1, wb + (size_t)16 * WSTRIDE, WSTRIDE, kk);
      wmma_acc(a00, fa0, fb0);
      wmma_acc(a01, fa0, fb1);
      wmma_acc(a10, fa1, fb0);
      wmma_acc(a11, fa1, fb1);
    }
    __syncthreads();
  }

  // epilogue: out[f][c] = (acc + b2[c] + skip(f,c)) * amask[f]
  {
    int nloc = lane & 15, mo = (lane >> 4) * 8;
    #pragma unroll
    for (int mi = 0; mi < 2; ++mi) {
      AccFrag* accA = mi ? &a10 : &a00;
      AccFrag* accB = mi ? &a11 : &a01;
      #pragma unroll
      for (int r = 0; r < 8; ++r) {
        int m = r + mo + mi * 16;
        int f = tileF * MT + m;
        float msk = amask[f];
        size_t xbase = (size_t)(f >> 3) * C + (size_t)(f & 7) * 16;
        int c0 = cbase + nloc, c1 = c0 + 16;
        float skip0 = x[xbase + (c0 >> 3)];
        float skip1 = x[xbase + (c1 >> 3)];
        out[(size_t)f * CO + c0] = (accA->f[r] + b2[c0] + skip0) * msk;
        out[(size_t)f * CO + c1] = (accB->f[r] + b2[c1] + skip1) * msk;
      }
    }
  }
}

// ---------------- host launcher ----------------
extern "C" void kernel_launch(void* const* d_in, const int* in_sizes, int n_in,
                              void* d_out, int out_size, void* d_ws, size_t ws_size,
                              hipStream_t stream) {
  const float* x      = (const float*)d_in[0];
  const int*   coords = (const int*)d_in[1];
  const float* gamma  = (const float*)d_in[2];
  const float* beta   = (const float*)d_in[3];
  const float* w_sub  = (const float*)d_in[4];
  const float* b_sub  = (const float*)d_in[5];
  const float* w1     = (const float*)d_in[6];
  const float* b1     = (const float*)d_in[7];
  const float* w2     = (const float*)d_in[8];
  const float* b2     = (const float*)d_in[9];

  float* out    = (float*)d_out;
  float* subdiv = out + (size_t)NF * CO;   // outputs concatenated: out[NF,CO] then subdiv[N,8]

  // workspace layout (bytes)
  char* ws = (char*)d_ws;
  int*      tc   = (int*)ws;                                 // 64^3 * 4     = 1,048,576
  int*      tf   = (int*)(ws + 1048576);                     // 128^3 * 4    = 8,388,608
  uint16_t* h_bf = (uint16_t*)(ws + 9437184);                // N*C*2        = 4,194,304
  float*    amsk = (float*)(ws + 13631488);                  // NF*4         =   524,288
  uint16_t* w1b  = (uint16_t*)(ws + 14155776);               // 1024*27*128*2= 7,077,888
  uint16_t* w2b  = (uint16_t*)(ws + 21233664);               // 128*27*128*2 =   884,736
  uint16_t* h2in = (uint16_t*)(ws + 22118400);               // NF*CO*2      = 33,554,432

  hipMemsetAsync(tc, 0xFF, (size_t)GD * GD * GD * 4, stream);   // tables -> -1
  hipMemsetAsync(tf, 0xFF, (size_t)GF * GF * GF * 4, stream);

  build_tables_k<<<(N + 127) / 128, 128, 0, stream>>>(coords, tc, tf);

  int nW1 = 1024 * 27 * 128, nW2 = 128 * 27 * 128;
  cvt_bf16_k<<<(nW1 + 255) / 256, 256, 0, stream>>>(w1, w1b, nW1);
  cvt_bf16_k<<<(nW2 + 255) / 256, 256, 0, stream>>>(w2, w2b, nW2);

  preprocess_k<<<N, 128, 0, stream>>>(x, gamma, beta, w_sub, b_sub, h_bf, amsk, subdiv);

  conv1_k<<<dim3(N / MT, 8), 128, 0, stream>>>(h_bf, coords, tc, w1b, b1, amsk, h2in);

  conv2_k<<<NF / MT, 128, 0, stream>>>(h2in, coords, tf, w2b, b2, amsk, x, out);
}
